// Decoder_37452114821320
// MI455X (gfx1250) — compile-verified, matrix operand order checked by reading, not run on hardware
//
#include <hip/hip_runtime.h>
#include <hip/hip_bf16.h>
#include <math.h>

typedef __bf16 bf16;
typedef __attribute__((ext_vector_type(16))) __bf16 v16bf;
typedef __attribute__((ext_vector_type(8)))  __bf16 v8bf;
typedef __attribute__((ext_vector_type(8)))  float  v8f;

#define B_  64
#define L_  128
#define E_  512
#define N_  512
#define H_  1024

static __device__ __forceinline__ v8bf ld8(const bf16* p){ return *(const v8bf*)p; }
static __device__ __forceinline__ v16bf mkfrag(v8bf lo, v8bf hi){
  union { v16bf v; v8bf h[2]; } u; u.h[0]=lo; u.h[1]=hi; return u.v;
}
static __device__ __forceinline__ v8f wmma_bf16(v16bf a, v16bf b, v8f c){
  return __builtin_amdgcn_wmma_f32_16x16x32_bf16(false, a, false, b, (short)0, c, false, false);
}
// One 16x16 f32 tile of A(MxK,row-major) * W(NoutxK,row-major)^T, K multiple of 32.
static __device__ __forceinline__ v8f wmma_tile(const bf16* arow, const bf16* wrow,
                                                int K, int half){
  v8f acc = {};
  for (int k0 = 0; k0 < K; k0 += 32){
    v16bf a = mkfrag(ld8(arow + k0 + 8*half), ld8(arow + k0 + 16 + 8*half));
    v16bf b = mkfrag(ld8(wrow + k0 + 8*half), ld8(wrow + k0 + 16 + 8*half));
    acc = wmma_bf16(a, b, acc);
  }
  return acc;
}
static __device__ __forceinline__ float sigmoidf_(float x){ return 1.f/(1.f+expf(-x)); }

// ---------------- setup kernels ----------------
__global__ void cvt_kernel(const float* __restrict__ src, bf16* __restrict__ dst, long n){
  long i = (long)blockIdx.x*blockDim.x + threadIdx.x;
  long stride = (long)gridDim.x*blockDim.x;
  for (; i < n; i += stride) dst[i] = (bf16)src[i];
}

__global__ void init_state_kernel(const float* __restrict__ s, float* __restrict__ st,
                                  bf16* __restrict__ hb, int n){
  int i = blockIdx.x*blockDim.x + threadIdx.x;
  if (i < n){ float v = s[i]; st[i] = v; hb[i] = (bf16)v; }
}

// keys = value @ Wk^T  (M = B*N = 32768, Nout = H, K = H), bf16 out.
// Each wave: 16x64 tile (4 accumulators share one A fragment).
__global__ void keys_kernel(const bf16* __restrict__ val, const bf16* __restrict__ Wk,
                            bf16* __restrict__ keys){
  int task = blockIdx.x*4 + (threadIdx.x >> 5);     // 2048 mt x 16 ng = 32768 tasks
  int lane = threadIdx.x & 31, half = lane >> 4, r = lane & 15;
  int mt = task >> 4, ng = task & 15;
  const bf16* arow = val + (size_t)(mt*16 + r) * H_;
  const bf16* w0 = Wk + (size_t)(ng*64 +  0 + r) * H_;
  const bf16* w1 = Wk + (size_t)(ng*64 + 16 + r) * H_;
  const bf16* w2 = Wk + (size_t)(ng*64 + 32 + r) * H_;
  const bf16* w3 = Wk + (size_t)(ng*64 + 48 + r) * H_;
  v8f c0 = {}, c1 = {}, c2 = {}, c3 = {};
  for (int k0 = 0; k0 < H_; k0 += 32){
    int a0 = k0 + 8*half, a1 = a0 + 16;
    v16bf a = mkfrag(ld8(arow + a0), ld8(arow + a1));
    c0 = wmma_bf16(a, mkfrag(ld8(w0 + a0), ld8(w0 + a1)), c0);
    c1 = wmma_bf16(a, mkfrag(ld8(w1 + a0), ld8(w1 + a1)), c1);
    c2 = wmma_bf16(a, mkfrag(ld8(w2 + a0), ld8(w2 + a1)), c2);
    c3 = wmma_bf16(a, mkfrag(ld8(w3 + a0), ld8(w3 + a1)), c3);
  }
  #pragma unroll
  for (int i = 0; i < 8; i++){
    size_t m = (size_t)(mt*16 + half*8 + i);
    int nb = ng*64 + r;
    keys[m*H_ + nb +  0] = (bf16)c0[i];
    keys[m*H_ + nb + 16] = (bf16)c1[i];
    keys[m*H_ + nb + 32] = (bf16)c2[i];
    keys[m*H_ + nb + 48] = (bf16)c3[i];
  }
}

// ---------------- per-step kernels ----------------
// Fused GRU gates: gx = [emb_t, h] @ Wih^T, gh = h @ Whh^T, full GRU cell,
// writes rnn_out (bf16) into cat[:, 0:H].
__global__ void gates_kernel(const bf16* __restrict__ emb, int t,
                             const bf16* __restrict__ hbf, const float* __restrict__ st,
                             const bf16* __restrict__ Wih, const bf16* __restrict__ Whh,
                             const float* __restrict__ b_ih, const float* __restrict__ b_hh,
                             bf16* __restrict__ cat){
  int tile = blockIdx.x*4 + (threadIdx.x >> 5);      // 4 x 64 = 256 tiles
  int lane = threadIdx.x & 31, half = lane >> 4, r = lane & 15;
  int mt = tile >> 6, nt = tile & 63;
  int am = mt*16 + r;
  const bf16* erow = emb + ((size_t)am*L_ + t) * E_;
  const bf16* hrow = hbf + (size_t)am * H_;
  int wr = nt*16 + r;
  const bf16* w0 = Wih + (size_t)(0*H_ + wr) * (E_+H_);
  const bf16* w1 = Wih + (size_t)(1*H_ + wr) * (E_+H_);
  const bf16* w2 = Wih + (size_t)(2*H_ + wr) * (E_+H_);
  v8f axr = {}, axz = {}, axn = {};
  for (int k0 = 0; k0 < E_+H_; k0 += 32){
    int c0 = k0 + 8*half, c1 = c0 + 16;
    v8bf alo = (c0 < E_) ? ld8(erow + c0) : ld8(hrow + (c0 - E_));
    v8bf ahi = (c1 < E_) ? ld8(erow + c1) : ld8(hrow + (c1 - E_));
    v16bf a  = mkfrag(alo, ahi);
    axr = wmma_bf16(a, mkfrag(ld8(w0 + c0), ld8(w0 + c1)), axr);
    axz = wmma_bf16(a, mkfrag(ld8(w1 + c0), ld8(w1 + c1)), axz);
    axn = wmma_bf16(a, mkfrag(ld8(w2 + c0), ld8(w2 + c1)), axn);
  }
  const bf16* v0 = Whh + (size_t)(0*H_ + wr) * H_;
  const bf16* v1 = Whh + (size_t)(1*H_ + wr) * H_;
  const bf16* v2 = Whh + (size_t)(2*H_ + wr) * H_;
  v8f ahr = {}, ahz = {}, ahn = {};
  for (int k0 = 0; k0 < H_; k0 += 32){
    int c0 = k0 + 8*half, c1 = c0 + 16;
    v16bf a  = mkfrag(ld8(hrow + c0), ld8(hrow + c1));
    ahr = wmma_bf16(a, mkfrag(ld8(v0 + c0), ld8(v0 + c1)), ahr);
    ahz = wmma_bf16(a, mkfrag(ld8(v1 + c0), ld8(v1 + c1)), ahz);
    ahn = wmma_bf16(a, mkfrag(ld8(v2 + c0), ld8(v2 + c1)), ahn);
  }
  int gcol = nt*16 + r;
  float bir = b_ih[gcol], biz = b_ih[H_+gcol], bin = b_ih[2*H_+gcol];
  float bhr = b_hh[gcol], bhz = b_hh[H_+gcol], bhn = b_hh[2*H_+gcol];
  #pragma unroll
  for (int i = 0; i < 8; i++){
    int m = mt*16 + half*8 + i;
    float rg = sigmoidf_(axr[i] + bir + ahr[i] + bhr);
    float zg = sigmoidf_(axz[i] + biz + ahz[i] + bhz);
    float nn = tanhf(axn[i] + bin + rg*(ahn[i] + bhn));
    float hp = st[(size_t)m*H_ + gcol];
    float out = (1.f - zg)*nn + zg*hp;
    cat[(size_t)m*(2*H_) + gcol] = (bf16)out;
  }
}

// q = rnn_out @ Wq^T  (rnn_out lives in cat[:,0:H], lda = 2H)
__global__ void qgemm_kernel(const bf16* __restrict__ cat, const bf16* __restrict__ Wq,
                             float* __restrict__ q){
  int tile = blockIdx.x*4 + (threadIdx.x >> 5);
  int lane = threadIdx.x & 31, half = lane >> 4, r = lane & 15;
  int mt = tile >> 6, nt = tile & 63;
  const bf16* arow = cat + (size_t)(mt*16 + r) * (2*H_);
  const bf16* wrow = Wq  + (size_t)(nt*16 + r) * H_;
  v8f acc = wmma_tile(arow, wrow, H_, half);
  int ncol = nt*16 + r;
  #pragma unroll
  for (int i = 0; i < 8; i++)
    q[(size_t)(mt*16 + half*8 + i)*H_ + ncol] = acc[i];
}

// Fused attention: score (v.tanh(q+keys)) -> masked softmax -> context.
// One 1024-thread block per batch element; q, v_att, score, attn live in LDS.
__global__ __launch_bounds__(1024)
void attn_kernel(const float* __restrict__ q, const bf16* __restrict__ keys,
                 const bf16* __restrict__ val, const float* __restrict__ vatt,
                 const float* __restrict__ mask, bf16* __restrict__ cat,
                 float* __restrict__ attn_out, int t){
  int b = blockIdx.x;
  int tid = threadIdx.x;                 // 0..1023
  int lane = tid & 31, wv = tid >> 5;    // 32 waves
  __shared__ float sq[H_], sv[H_], sc[N_], red[N_];
  sq[tid] = q[(size_t)b*H_ + tid];
  sv[tid] = vatt[tid];
  __syncthreads();
  // scores: wave wv handles rows n = wv*16 .. wv*16+15
  for (int i = 0; i < 16; ++i){
    int n = wv*16 + i;
    const bf16* krow = keys + ((size_t)b*N_ + n) * H_;
    float acc = 0.f;
    #pragma unroll
    for (int it = 0; it < 4; ++it){
      int h = it*256 + lane*8;
      v8bf kv = ld8(krow + h);
      #pragma unroll
      for (int j = 0; j < 8; j++)
        acc += sv[h+j] * tanhf(sq[h+j] + (float)kv[j]);
    }
    #pragma unroll
    for (int off = 16; off > 0; off >>= 1) acc += __shfl_xor(acc, off, 32);
    if (lane == 0)
      sc[n] = (mask[(size_t)b*N_ + n] > 0.f) ? acc : -1e9f;
  }
  __syncthreads();
  // softmax over sc[0..511]
  if (tid < N_) red[tid] = sc[tid];
  __syncthreads();
  for (int off = N_/2; off > 0; off >>= 1){
    if (tid < off) red[tid] = fmaxf(red[tid], red[tid+off]);
    __syncthreads();
  }
  float mx = red[0];
  __syncthreads();
  float e = 0.f;
  if (tid < N_){ e = expf(sc[tid] - mx); red[tid] = e; }
  __syncthreads();
  for (int off = N_/2; off > 0; off >>= 1){
    if (tid < off) red[tid] += red[tid+off];
    __syncthreads();
  }
  float inv = 1.f / red[0];
  __syncthreads();
  if (tid < N_){
    float a = e * inv;
    sc[tid] = a;
    attn_out[((size_t)b*L_ + t)*N_ + tid] = a;
  }
  __syncthreads();
  // context[b, h=tid] = sum_n attn[n]*value[b,n,h] -> cat[:, H:2H] (bf16)
  float acc = 0.f;
  const bf16* vb = val + (size_t)b*N_*H_ + tid;
  for (int n = 0; n < N_; ++n) acc += sc[n] * (float)vb[(size_t)n*H_];
  cat[(size_t)b*(2*H_) + H_ + tid] = (bf16)acc;
}

// new_state = tanh(cat @ Wm^T + bm); writes outputs[b,t,:], state_f32, h_bf16
__global__ void mlp_kernel(const bf16* __restrict__ cat, const bf16* __restrict__ Wm,
                           const float* __restrict__ bm, float* __restrict__ outs, int t,
                           float* __restrict__ st, bf16* __restrict__ hbf){
  int tile = blockIdx.x*4 + (threadIdx.x >> 5);
  int lane = threadIdx.x & 31, half = lane >> 4, r = lane & 15;
  int mt = tile >> 6, nt = tile & 63;
  const bf16* arow = cat + (size_t)(mt*16 + r) * (2*H_);
  const bf16* wrow = Wm  + (size_t)(nt*16 + r) * (2*H_);
  v8f acc = wmma_tile(arow, wrow, 2*H_, half);
  int ncol = nt*16 + r;
  float bv = bm[ncol];
  #pragma unroll
  for (int i = 0; i < 8; i++){
    int m = mt*16 + half*8 + i;
    float v = tanhf(acc[i] + bv);
    outs[((size_t)m*L_ + t)*H_ + ncol] = v;
    st[(size_t)m*H_ + ncol] = v;
    hbf[(size_t)m*H_ + ncol] = (bf16)v;
  }
}

extern "C" void kernel_launch(void* const* d_in, const int* in_sizes, int n_in,
                              void* d_out, int out_size, void* d_ws, size_t ws_size,
                              hipStream_t stream){
  (void)in_sizes; (void)n_in; (void)out_size; (void)ws_size;
  const float* emb_f = (const float*)d_in[0];
  const float* val_f = (const float*)d_in[1];
  const float* mask  = (const float*)d_in[2];
  const float* state = (const float*)d_in[3];
  const float* Wih_f = (const float*)d_in[4];
  const float* bih   = (const float*)d_in[5];
  const float* Whh_f = (const float*)d_in[6];
  const float* bhh   = (const float*)d_in[7];
  const float* Wq_f  = (const float*)d_in[8];
  const float* Wk_f  = (const float*)d_in[9];
  const float* vatt  = (const float*)d_in[10];
  const float* Wm_f  = (const float*)d_in[11];
  const float* bm    = (const float*)d_in[12];

  float* outs     = (float*)d_out;                        // B*L*H
  float* attn_out = outs + (size_t)B_*L_*H_;              // B*L*N

  char* w = (char*)d_ws;
  auto alloc = [&](size_t bytes){ char* p = w; w += (bytes + 255) & ~(size_t)255; return p; };
  bf16*  hWih = (bf16*)alloc((size_t)3*H_*(E_+H_)*2);
  bf16*  hWhh = (bf16*)alloc((size_t)3*H_*H_*2);
  bf16*  hWq  = (bf16*)alloc((size_t)H_*H_*2);
  bf16*  hWk  = (bf16*)alloc((size_t)H_*H_*2);
  bf16*  hWm  = (bf16*)alloc((size_t)H_*2*H_*2);
  bf16*  hVal = (bf16*)alloc((size_t)B_*N_*H_*2);
  bf16*  hKey = (bf16*)alloc((size_t)B_*N_*H_*2);
  bf16*  hEmb = (bf16*)alloc((size_t)B_*L_*E_*2);
  float* st32 = (float*)alloc((size_t)B_*H_*4);
  bf16*  hbf  = (bf16*)alloc((size_t)B_*H_*2);
  bf16*  cat  = (bf16*)alloc((size_t)B_*2*H_*2);
  float* q    = (float*)alloc((size_t)B_*H_*4);

  dim3 cb(256);
  cvt_kernel<<<1024, cb, 0, stream>>>(Wih_f, hWih, (long)3*H_*(E_+H_));
  cvt_kernel<<<1024, cb, 0, stream>>>(Whh_f, hWhh, (long)3*H_*H_);
  cvt_kernel<<<512,  cb, 0, stream>>>(Wq_f,  hWq,  (long)H_*H_);
  cvt_kernel<<<512,  cb, 0, stream>>>(Wk_f,  hWk,  (long)H_*H_);
  cvt_kernel<<<512,  cb, 0, stream>>>(Wm_f,  hWm,  (long)H_*2*H_);
  cvt_kernel<<<2048, cb, 0, stream>>>(val_f, hVal, (long)B_*N_*H_);
  cvt_kernel<<<1024, cb, 0, stream>>>(emb_f, hEmb, (long)B_*L_*E_);
  init_state_kernel<<<(B_*H_ + 255)/256, cb, 0, stream>>>(state, st32, hbf, B_*H_);
  keys_kernel<<<8192, dim3(128), 0, stream>>>(hVal, hWk, hKey);

  for (int t = 0; t < L_; ++t){
    gates_kernel<<<64,  dim3(128),  0, stream>>>(hEmb, t, hbf, st32, hWih, hWhh, bih, bhh, cat);
    qgemm_kernel<<<64,  dim3(128),  0, stream>>>(cat, hWq, q);
    attn_kernel <<<64,  dim3(1024), 0, stream>>>(q, hKey, hVal, vatt, mask, cat, attn_out, t);
    mlp_kernel  <<<64,  dim3(128),  0, stream>>>(cat, hWm, bm, outs, t, st32, hbf);
  }
}
